// MSDeformAttn_16922171146928
// MI455X (gfx1250) — matrix operand order repaired and checked by reading, not compiled, not run on hardware
//
#include <hip/hip_runtime.h>
#include <hip/hip_bf16.h>

// ---------------------------------------------------------------------------
// MS-Deformable-Attention for gfx1250 (MI455X), wave32 + bf16 WMMA.
// Pipeline: cvt/pack (once) -> 3 WMMA GEMMs -> softmax -> bilinear gather
//           (bf16 out) -> final WMMA GEMM.
// Problem constants: N=2, Lq=22000, Len_in=22000, D_MODEL=256, D_HEAD=64,
//   H=8, L=2, P=4, shapes {(100,176),(50,88)}, level starts {0,17600}.
// ---------------------------------------------------------------------------

#define NBATCH   2
#define LQ       22000
#define LEN_IN   22000
#define DMODEL   256
#define DHEAD    64
#define NHEADS   8
#define HD       (NHEADS * DHEAD)   // 512
#define ROWS     (NBATCH * LQ)      // 44000 (= 32 * 1375)

typedef __attribute__((ext_vector_type(16))) __bf16          v16bf;
typedef __attribute__((ext_vector_type(16))) unsigned short  v16u16;
typedef __attribute__((ext_vector_type(8)))  float           v8f;

union Frag {                     // 32B fragment = 16 bf16 halves = 2x uint4
    v16u16 v;
    uint4  q[2];
};

// round-to-nearest-even f32 -> bf16 bit pattern
static __device__ __forceinline__ unsigned short bf16bits(float f) {
    unsigned u = __builtin_bit_cast(unsigned, f);
    unsigned r = u + 0x7FFFu + ((u >> 16) & 1u);
    return (unsigned short)(r >> 16);
}
static __device__ __forceinline__ unsigned pk_bf16(float lo, float hi) {
    return (unsigned)bf16bits(lo) | ((unsigned)bf16bits(hi) << 16);
}

// ---------------------------------------------------------------------------
// f32 -> bf16 bulk convert, 8 elements / thread, b128 in / b128 out.
// ---------------------------------------------------------------------------
__global__ void cvt_bf16_kernel(const float* __restrict__ src,
                                unsigned short* __restrict__ dst, long n8) {
    const long t = (long)blockIdx.x * blockDim.x + threadIdx.x;
    if (t >= n8) return;
    const float4* s = (const float4*)src + t * 2;
    float4 a = s[0], b = s[1];
    ((uint4*)dst)[t] = make_uint4(pk_bf16(a.x, a.y), pk_bf16(a.z, a.w),
                                  pk_bf16(b.x, b.y), pk_bf16(b.z, b.w));
}

// ---------------------------------------------------------------------------
// Pack f32 weight B[K][Nc] (row-major) into bf16 WMMA B-fragments, kc-major:
//   Bp[(((kc*TILESN + tn)*32) + lane)*16 + j]
//   lane 0-15 : col = tn*16+lane,    j -> K = kc*32 + j
//   lane16-31 : col = tn*16+lane-16, j -> K = kc*32 + 16 + j
// ---------------------------------------------------------------------------
__global__ void pack_b_kernel(const float* __restrict__ B,
                              unsigned short* __restrict__ Bp,
                              int K, int Nc) {
    const int tilesN = Nc >> 4;
    const long total = (long)(K >> 5) * tilesN * 32;
    const long t = (long)blockIdx.x * blockDim.x + threadIdx.x;
    if (t >= total) return;
    const int lane = (int)(t & 31);
    const int tn   = (int)((t >> 5) % tilesN);
    const int kc   = (int)((t >> 5) / tilesN);
    const int col   = tn * 16 + (lane & 15);
    const int kbase = kc * 32 + ((lane >> 4) ? 16 : 0);
    unsigned short* o = Bp + t * 16;
#pragma unroll
    for (int j = 0; j < 16; ++j)
        o[j] = bf16bits(B[(long)(kbase + j) * Nc + col]);
}

// ---------------------------------------------------------------------------
// C[M,Nc] = A[M,K](bf16) @ Bp(packed bf16 fragments) + bias, C f32.
// Compile-time KCC = K/32, TILESN = Nc/16 -> every stride is an immediate.
// One wave computes 32x64: 2 M-tiles x 4 N-tiles = 8 accumulators.
// Steady-state per k-step: 4 A b128 + 8 B b128 (one clause, const offsets)
// + 8 independent v_wmma_f32_16x16x32_bf16.
// M mult of 32; Nc mult of 64.
// ---------------------------------------------------------------------------
template <int KCC, int TILESN>
__global__ void wmma_gemm_bias(const unsigned short* __restrict__ A,
                               const unsigned short* __restrict__ Bp,
                               const float* __restrict__ bias,
                               float* __restrict__ C,
                               int M) {
    constexpr int K  = KCC * 32;
    constexpr int Nc = TILESN * 16;
    constexpr int groupsN = TILESN / 4;

    const int lane = threadIdx.x & 31;
    const int wave = threadIdx.x >> 5;
    const long total = (long)(M >> 5) * groupsN;
    const long t = (long)blockIdx.x * (blockDim.x >> 5) + wave;
    if (t >= total) return;                       // wave-uniform branch

    const int gN = (int)(t % groupsN);
    const int gM = (int)(t / groupsN);
    const int tileN0 = gN * 4;
    const int l15 = lane & 15;
    const int hi  = lane >> 4;

    // A row pointers for the two M-tiles
    const unsigned short* __restrict__ arow0 =
        A + (long)(gM * 32 + l15) * K + (hi ? 16 : 0);
    const unsigned short* __restrict__ arow1 = arow0 + 16 * K;

    // B fragment base for this lane / N-group (kc = 0)
    const uint4* __restrict__ bbase =
        (const uint4*)Bp + ((long)tileN0 * 32 + lane) * 2;

    v8f acc[2][4];
#pragma unroll
    for (int m = 0; m < 2; ++m)
#pragma unroll
        for (int s = 0; s < 4; ++s) acc[m][s] = {};

    for (int kc = 0; kc < KCC; ++kc) {
        // A fragments: elems 0..15 = K[ka..ka+15]
        const uint4* ap0 = (const uint4*)(arow0 + kc * 32);
        const uint4* ap1 = (const uint4*)(arow1 + kc * 32);
        Frag a0; a0.q[0] = ap0[0]; a0.q[1] = ap0[1];
        Frag a1; a1.q[0] = ap1[0]; a1.q[1] = ap1[1];
        v16bf af[2] = { __builtin_bit_cast(v16bf, a0.v),
                        __builtin_bit_cast(v16bf, a1.v) };

        // 4 B fragments: contiguous 4KB block, constant offsets
        const uint4* bp = bbase + (long)kc * (TILESN * 64);
        v16bf bf[4];
#pragma unroll
        for (int s = 0; s < 4; ++s) {
            Frag b; b.q[0] = bp[s * 64]; b.q[1] = bp[s * 64 + 1];
            bf[s] = __builtin_bit_cast(v16bf, b.v);
        }

#pragma unroll
        for (int m = 0; m < 2; ++m)
#pragma unroll
            for (int s = 0; s < 4; ++s)
                acc[m][s] = __builtin_amdgcn_wmma_f32_16x16x32_bf16(
                    false, af[m], false, bf[s], (short)0, acc[m][s],
                    false, false);
    }

    // C/D layout: VGPR r -> row base + r + hi*8, col = tile base + l15
#pragma unroll
    for (int m = 0; m < 2; ++m) {
        const int mbase = gM * 32 + m * 16 + (hi ? 8 : 0);
#pragma unroll
        for (int s = 0; s < 4; ++s) {
            const int ncol = (tileN0 + s) * 16 + l15;
            const float bval = bias[ncol];
#pragma unroll
            for (int r = 0; r < 8; ++r)
                C[(long)(mbase + r) * Nc + ncol] = acc[m][s][r] + bval;
        }
    }
}

// ---------------------------------------------------------------------------
// In-place softmax over groups of 8 (L*P) within each (row, head).
// attn layout: [ROWS][64], cols = h*8 + lp.
// ---------------------------------------------------------------------------
__global__ void softmax8_kernel(float* __restrict__ attn) {
    const long t = (long)blockIdx.x * blockDim.x + threadIdx.x;
    if (t >= (long)ROWS * NHEADS) return;
    float* p = attn + (t >> 3) * 64 + (t & 7) * 8;
    float v[8], m = -3.4e38f;
#pragma unroll
    for (int i = 0; i < 8; ++i) { v[i] = p[i]; m = fmaxf(m, v[i]); }
    float s = 0.f;
#pragma unroll
    for (int i = 0; i < 8; ++i) { v[i] = __expf(v[i] - m); s += v[i]; }
    const float inv = 1.f / s;
#pragma unroll
    for (int i = 0; i < 8; ++i) p[i] = v[i] * inv;
}

// ---------------------------------------------------------------------------
// Deformable bilinear gather. One wave per (n,q,h); lane owns channels
// d = 2*lane, 2*lane+1 -> float2 gathers (256B coalesced per corner).
// value is L2-resident (90MB < 192MB L2). Writes core directly as bf16.
// Note: (ref+off)/norm * W - 0.5 reduces to ref+off-0.5 (norm == (Wl,Hl)).
// ---------------------------------------------------------------------------
__global__ void deform_core_kernel(const float* __restrict__ value,
                                   const float* __restrict__ off,
                                   const float* __restrict__ attn,
                                   const float* __restrict__ refp,
                                   unsigned short* __restrict__ core) {
    const int lane = threadIdx.x & 31;
    const int wave = threadIdx.x >> 5;
    const long w = (long)blockIdx.x * (blockDim.x >> 5) + wave;
    if (w >= (long)ROWS * NHEADS) return;

    const int h = (int)(w % NHEADS);
    const long rowq = w / NHEADS;                 // n*LQ + q
    const int n = (int)(rowq / LQ);

    const int Hls[2]    = {100, 50};
    const int Wls[2]    = {176, 88};
    const int starts[2] = {0, 17600};

    const float* __restrict__ offp = off  + rowq * 128 + h * 16;
    const float* __restrict__ awp  = attn + rowq * 64  + h * 8;
    const int d = lane * 2;
    float acc0 = 0.f, acc1 = 0.f;

#pragma unroll
    for (int l = 0; l < 2; ++l) {
        const int Wi = Wls[l], Hi = Hls[l];
        const float rx = refp[(rowq * 2 + l) * 2 + 0];
        const float ry = refp[(rowq * 2 + l) * 2 + 1];
        const float* __restrict__ vbase =
            value + ((long)n * LEN_IN + starts[l]) * HD + h * DHEAD + d;
#pragma unroll
        for (int p = 0; p < 4; ++p) {
            const float x  = rx + offp[l * 8 + p * 2 + 0] - 0.5f;
            const float y  = ry + offp[l * 8 + p * 2 + 1] - 0.5f;
            const float aw = awp[l * 4 + p];
            const float x0f = floorf(x), y0f = floorf(y);
            const float fx = x - x0f, fy = y - y0f;
            const int x0 = (int)x0f, y0 = (int)y0f;
#pragma unroll
            for (int c = 0; c < 4; ++c) {
                const int dx = c & 1, dy = c >> 1;
                const int xi = x0 + dx, yi = y0 + dy;
                const bool valid = (xi >= 0) & (xi < Wi) & (yi >= 0) & (yi < Hi);
                const float wgt = valid
                    ? (dx ? fx : 1.f - fx) * (dy ? fy : 1.f - fy) * aw : 0.f;
                const int xc = min(max(xi, 0), Wi - 1);
                const int yc = min(max(yi, 0), Hi - 1);
                const float2 g =
                    *(const float2*)(vbase + (long)(yc * Wi + xc) * HD);
                acc0 += wgt * g.x;
                acc1 += wgt * g.y;
            }
        }
    }
    *(unsigned*)(core + rowq * HD + h * DHEAD + d) = pk_bf16(acc0, acc1);
}

// ---------------------------------------------------------------------------
// Host launcher
// ---------------------------------------------------------------------------
extern "C" void kernel_launch(void* const* d_in, const int* in_sizes, int n_in,
                              void* d_out, int out_size, void* d_ws, size_t ws_size,
                              hipStream_t stream) {
    (void)in_sizes; (void)n_in; (void)out_size; (void)ws_size;

    const float* query  = (const float*)d_in[0];   // [2,22000,256]
    const float* refp   = (const float*)d_in[1];   // [2,22000,2,2]
    const float* inflat = (const float*)d_in[2];   // [2,22000,256]
    const float* Wv     = (const float*)d_in[3];   // [256,512]
    const float* bv     = (const float*)d_in[4];   // [512]
    const float* Woff   = (const float*)d_in[5];   // [256,128]
    const float* boff   = (const float*)d_in[6];   // [128]
    const float* Wattn  = (const float*)d_in[7];   // [256,64]
    const float* battn  = (const float*)d_in[8];   // [64]
    const float* Wo     = (const float*)d_in[9];   // [512,256]
    const float* bo     = (const float*)d_in[10];  // [256]
    float* out = (float*)d_out;                    // [2,22000,256]

    // ---- workspace carve-up (all segments stay 16B aligned) ----
    float* value = (float*)d_ws;                               // ROWS*512 f32
    float* off   = value + (long)ROWS * HD;                    // ROWS*128 f32
    float* attn  = off   + (long)ROWS * 128;                   // ROWS*64  f32
    unsigned short* qb    = (unsigned short*)(attn + (long)ROWS * 64);
    unsigned short* inb   = qb    + (long)ROWS * DMODEL;       // bf16 acts
    unsigned short* coreb = inb   + (long)ROWS * DMODEL;       // ROWS*512 bf16
    unsigned short* pWv   = coreb + (long)ROWS * HD;           // packed weights
    unsigned short* pWoff = pWv   + (long)(HD / 16) * (DMODEL / 32) * 512;
    unsigned short* pWat  = pWoff + (long)(128 / 16) * (DMODEL / 32) * 512;
    unsigned short* pWo   = pWat  + (long)(64 / 16) * (DMODEL / 32) * 512;

    const int THREADS = 256;
    const int WPB = THREADS / 32;

    // ---- one-time converts / packs (tiny or memory-bound single pass) ----
    {
        long n8 = (long)ROWS * DMODEL / 8;
        int blocks = (int)((n8 + THREADS - 1) / THREADS);
        cvt_bf16_kernel<<<blocks, THREADS, 0, stream>>>(query,  qb,  n8);
        cvt_bf16_kernel<<<blocks, THREADS, 0, stream>>>(inflat, inb, n8);
    }
    auto pack = [&](const float* B, unsigned short* Bp, int K, int Nc) {
        long tthreads = (long)(K / 32) * (Nc / 16) * 32;
        int blocks = (int)((tthreads + THREADS - 1) / THREADS);
        pack_b_kernel<<<blocks, THREADS, 0, stream>>>(B, Bp, K, Nc);
    };
    pack(Wv,    pWv,   DMODEL, HD);
    pack(Woff,  pWoff, DMODEL, 128);
    pack(Wattn, pWat,  DMODEL, 64);
    pack(Wo,    pWo,   HD,     DMODEL);

    auto gemm_blocks = [&](int M, int Nc) {
        long groups = (long)(M / 32) * (Nc / 64);
        return (int)((groups + WPB - 1) / WPB);
    };

    // 1) value = input_flatten @ Wv + bv            (44000 x 256 x 512)
    wmma_gemm_bias<8, 32><<<gemm_blocks(ROWS, HD), THREADS, 0, stream>>>(
        inb, pWv, bv, value, ROWS);

    // 2) off = query @ Woff + boff                  (44000 x 256 x 128)
    wmma_gemm_bias<8, 8><<<gemm_blocks(ROWS, 128), THREADS, 0, stream>>>(
        qb, pWoff, boff, off, ROWS);

    // 3) attn = query @ Wattn + battn               (44000 x 256 x 64)
    wmma_gemm_bias<8, 4><<<gemm_blocks(ROWS, 64), THREADS, 0, stream>>>(
        qb, pWat, battn, attn, ROWS);

    // 4) softmax over the 8 (level,point) weights per (row, head)
    {
        long t = (long)ROWS * NHEADS;
        int blocks = (int)((t + THREADS - 1) / THREADS);
        softmax8_kernel<<<blocks, THREADS, 0, stream>>>(attn);
    }

    // 5) deformable bilinear gather -> core (bf16)
    {
        long waves = (long)ROWS * NHEADS;
        int blocks = (int)((waves + WPB - 1) / WPB);
        deform_core_kernel<<<blocks, THREADS, 0, stream>>>(
            value, off, attn, refp, coreb);
    }

    // 6) out = core @ Wo + bo                       (44000 x 512 x 256)
    wmma_gemm_bias<16, 16><<<gemm_blocks(ROWS, DMODEL), THREADS, 0, stream>>>(
        coreb, pWo, bo, out, ROWS);
}
